// Embedder_76527727280252
// MI455X (gfx1250) — matrix operand order repaired and checked by reading, not compiled, not run on hardware
//
#include <hip/hip_runtime.h>
#include <stdint.h>

typedef __attribute__((ext_vector_type(16))) __bf16 v16bf;
typedef __attribute__((ext_vector_type(8)))  __bf16 v8bf;
typedef __attribute__((ext_vector_type(8)))  float  v8f;

#define N_NODES 100000
#define N_RELS  19

__device__ __forceinline__ v8f wmma_bf16(v16bf a, v16bf b, v8f c) {
  // D = A(16x32 bf16) x B(32x16 bf16) + C(16x16 f32)
  return __builtin_amdgcn_wmma_f32_16x16x32_bf16(
      /*neg_a=*/false, a, /*neg_b=*/false, b,
      /*c_mod=*/(short)0, c, /*reuse_a=*/false, /*reuse_b=*/false);
}

// ---------------- setup kernels ----------------

__global__ void k_zero(uint32_t* p, long long nwords) {
  long long i = (long long)blockIdx.x * blockDim.x + threadIdx.x;
  if (i < nwords) p[i] = 0u;
}

__global__ void k_count(const int* __restrict__ dst, const int* __restrict__ et,
                        float* __restrict__ deg, uint32_t* __restrict__ cnt, int E) {
  int e = blockIdx.x * blockDim.x + threadIdx.x;
  if (e < E) {
    atomicAdd(&deg[dst[e]], 1.0f);
    atomicAdd(&cnt[et[e]], 1u);
  }
}

__global__ void k_prefix(const uint32_t* __restrict__ cnt, uint32_t* rel_off,
                         uint32_t* cursor, uint32_t* btoff) {
  if (threadIdx.x == 0) {
    uint32_t ro = 0, bo = 0;
    for (int r = 0; r < N_RELS; ++r) {
      rel_off[r] = ro; cursor[r] = ro; btoff[r] = bo;
      ro += cnt[r];
      bo += (cnt[r] + 63u) >> 6;       // 64-edge block tiles per relation
    }
    rel_off[N_RELS] = ro; btoff[N_RELS] = bo;
  }
}

__global__ void k_scatter(const int* __restrict__ et, uint32_t* __restrict__ cursor,
                          uint32_t* __restrict__ perm, int E) {
  int e = blockIdx.x * blockDim.x + threadIdx.x;
  if (e < E) {
    uint32_t slot = atomicAdd(&cursor[et[e]], 1u);
    perm[slot] = (uint32_t)e;
  }
}

__global__ void k_segS(const int* __restrict__ src, const int* __restrict__ dst,
                       const int* __restrict__ et, const float* __restrict__ deg,
                       float* __restrict__ S, int E) {
  int e = blockIdx.x * blockDim.x + threadIdx.x;
  if (e < E) atomicAdd(&S[(long long)dst[e] * N_RELS + et[e]], deg[src[e]]);
}

// Layer 0 (din==1 collapses): h1[v,c] = relu(b0[c] + deg[v]*Wl0[c] + sum_r S[v,r]*W0[r,c])
__global__ void k_layer0(const float* __restrict__ S, const float* __restrict__ deg,
                         const float* __restrict__ W0, const float* __restrict__ Wl0,
                         const float* __restrict__ b0, __bf16* __restrict__ hOut) {
  int v = blockIdx.x;          // N blocks
  int c = threadIdx.x;         // 128 threads
  float f = b0[c] + deg[v] * Wl0[c];
  const float* srow = S + (long long)v * N_RELS;
#pragma unroll
  for (int r = 0; r < N_RELS; ++r) f += srow[r] * W0[r * 128 + c];
  f = fmaxf(f, 0.0f);
  hOut[(long long)v * 128 + c] = (__bf16)f;
}

__global__ void k_relu_cvt(const float* __restrict__ acc, __bf16* __restrict__ hOut,
                           long long n) {
  long long i = (long long)blockIdx.x * blockDim.x + threadIdx.x;
  if (i < n) hOut[i] = (__bf16)fmaxf(acc[i], 0.0f);
}

// ---------------- WMMA GEMM kernels ----------------
// Self-loop: out[v,:] = h[v,:] @ Wloop + b. 4 waves/block, one 16-node tile per wave.
template <int DIN, int DOUT>
__global__ void k_selfloop(const __bf16* __restrict__ hIn, const float* __restrict__ Wl,
                           const float* __restrict__ bias, float* __restrict__ out,
                           int nTiles) {
  __shared__ __bf16 Wt[DOUT * DIN];                 // transposed: Wt[n*DIN + k], bf16
  for (int idx = threadIdx.x; idx < DIN * DOUT; idx += blockDim.x) {
    int k = idx / DOUT, n = idx % DOUT;
    Wt[n * DIN + k] = (__bf16)Wl[idx];
  }
  __syncthreads();

  int wave = threadIdx.x >> 5, lane = threadIdx.x & 31;
  int tile = blockIdx.x * 4 + wave;
  if (tile >= nTiles) return;
  int g = lane >> 4, l16 = lane & 15;

  constexpr int NT = DOUT / 16;
  const v8f vzero = {0.f, 0.f, 0.f, 0.f, 0.f, 0.f, 0.f, 0.f};
  v8f acc[NT];
#pragma unroll
  for (int t = 0; t < NT; ++t) acc[t] = vzero;

  const __bf16* rowPtr = hIn + (long long)(tile * 16 + l16) * DIN + g * 8;
#pragma unroll
  for (int kc = 0; kc < DIN / 32; ++kc) {
    int kb = kc * 32;
    v8bf alo = *(const v8bf*)(rowPtr + kb);         // K = kb+8g .. +8
    v8bf ahi = *(const v8bf*)(rowPtr + kb + 16);    // K = kb+16+8g .. +8
    v16bf A;
#pragma unroll
    for (int i = 0; i < 8; ++i) { A[i] = alo[i]; A[8 + i] = ahi[i]; }
#pragma unroll
    for (int t = 0; t < NT; ++t) {
      const __bf16* bp = &Wt[(t * 16 + l16) * DIN + kb + g * 8];  // K = kb+8g..+8, kb+16+8g..+8
      v8bf blo = *(const v8bf*)bp;
      v8bf bhi = *(const v8bf*)(bp + 16);
      v16bf B;
#pragma unroll
      for (int i = 0; i < 8; ++i) { B[i] = blo[i]; B[8 + i] = bhi[i]; }
      acc[t] = wmma_bf16(A, B, acc[t]);
    }
  }
#pragma unroll
  for (int t = 0; t < NT; ++t) {
    float bv = bias[t * 16 + l16];
#pragma unroll
    for (int j = 0; j < 8; ++j) {
      int M = j + 8 * g;                            // C layout: VGPR j, lane-group g
      out[(long long)(tile * 16 + M) * DOUT + t * 16 + l16] = acc[t][j] + bv;
    }
  }
}

// Edge GEMM: edges relation-sorted in perm. Block = 4 waves, 64 edges, one relation;
// W[r] staged to LDS (bf16, transposed). A = gathered h[src] rows; scatter-add by dst.
template <int DIN, int DOUT>
__global__ void k_edge(const __bf16* __restrict__ hIn, const float* __restrict__ W,
                       const int* __restrict__ srcA, const int* __restrict__ dstA,
                       const uint32_t* __restrict__ perm,
                       const uint32_t* __restrict__ rel_off,
                       const uint32_t* __restrict__ btoff,
                       float* __restrict__ out) {
  uint32_t nblk = btoff[N_RELS];
  if (blockIdx.x >= nblk) return;
  int r = 0;
  while (btoff[r + 1] <= blockIdx.x) ++r;           // <=19 scalar iterations

  __shared__ __bf16 Wt[DOUT * DIN];
  const float* Wr = W + (long long)r * DIN * DOUT;
  for (int idx = threadIdx.x; idx < DIN * DOUT; idx += blockDim.x) {
    int k = idx / DOUT, n = idx % DOUT;
    Wt[n * DIN + k] = (__bf16)Wr[idx];
  }
  __syncthreads();

  int wave = threadIdx.x >> 5, lane = threadIdx.x & 31;
  long long tileStart = (long long)rel_off[r] +
                        (long long)(blockIdx.x - btoff[r]) * 64 + (long long)wave * 16;
  long long relEnd = rel_off[r + 1];
  long long rem = relEnd - tileStart;
  int cnt = rem >= 16 ? 16 : (int)rem;
  if (cnt <= 0) return;                             // wave-uniform
  int g = lane >> 4, l16 = lane & 15;

  // A row gather (pad lanes clamp to a valid edge; their output rows are discarded)
  int eLocal = l16 < cnt - 1 ? l16 : cnt - 1;
  int srcRow = srcA[perm[tileStart + eLocal]];
  const __bf16* rowPtr = hIn + (long long)srcRow * DIN + g * 8;

  int drow[8];
#pragma unroll
  for (int j = 0; j < 8; ++j) {
    int M = j + 8 * g;
    drow[j] = (M < cnt) ? dstA[perm[tileStart + M]] : -1;
  }

  constexpr int NT = DOUT / 16;
  const v8f vzero = {0.f, 0.f, 0.f, 0.f, 0.f, 0.f, 0.f, 0.f};
  v8f acc[NT];
#pragma unroll
  for (int t = 0; t < NT; ++t) acc[t] = vzero;

#pragma unroll
  for (int kc = 0; kc < DIN / 32; ++kc) {
    int kb = kc * 32;
    v8bf alo = *(const v8bf*)(rowPtr + kb);
    v8bf ahi = *(const v8bf*)(rowPtr + kb + 16);
    v16bf A;
#pragma unroll
    for (int i = 0; i < 8; ++i) { A[i] = alo[i]; A[8 + i] = ahi[i]; }
#pragma unroll
    for (int t = 0; t < NT; ++t) {
      const __bf16* bp = &Wt[(t * 16 + l16) * DIN + kb + g * 8];
      v8bf blo = *(const v8bf*)bp;
      v8bf bhi = *(const v8bf*)(bp + 16);
      v16bf B;
#pragma unroll
      for (int i = 0; i < 8; ++i) { B[i] = blo[i]; B[8 + i] = bhi[i]; }
      acc[t] = wmma_bf16(A, B, acc[t]);
    }
  }
#pragma unroll
  for (int t = 0; t < NT; ++t) {
    int col = t * 16 + l16;
#pragma unroll
    for (int j = 0; j < 8; ++j) {
      if (drow[j] >= 0)
        atomicAdd(&out[(long long)drow[j] * DOUT + col], acc[t][j]);
    }
  }
}

// ---------------- launch ----------------

extern "C" void kernel_launch(void* const* d_in, const int* in_sizes, int n_in,
                              void* d_out, int out_size, void* d_ws, size_t ws_size,
                              hipStream_t stream) {
  const int*   src = (const int*)d_in[0];
  const int*   dst = (const int*)d_in[1];
  const int*   et  = (const int*)d_in[2];
  const float* W0  = (const float*)d_in[3];
  const float* Wl0 = (const float*)d_in[4];
  const float* b0  = (const float*)d_in[5];
  const float* W1  = (const float*)d_in[6];
  const float* Wl1 = (const float*)d_in[7];
  const float* b1  = (const float*)d_in[8];
  const float* W2  = (const float*)d_in[9];
  const float* Wl2 = (const float*)d_in[10];
  const float* b2  = (const float*)d_in[11];
  const int E = in_sizes[0];
  const int N = N_NODES;
  (void)n_in; (void)out_size; (void)ws_size;

  char* ws = (char*)d_ws;
  size_t off = 0;
  auto take = [&](size_t bytes) {
    size_t o = off;
    off += (bytes + 255) & ~(size_t)255;
    return o;
  };
  size_t o_deg  = take((size_t)N * 4);
  size_t o_S    = take((size_t)N * N_RELS * 4);
  size_t o_cnt  = take((size_t)N_RELS * 4);
  size_t zeroEnd = off;                       // zero deg, S, cnt
  size_t o_cur  = take((size_t)N_RELS * 4);
  size_t o_ro   = take((size_t)(N_RELS + 1) * 4);
  size_t o_bt   = take((size_t)(N_RELS + 1) * 4);
  size_t o_perm = take((size_t)E * 4);
  size_t o_hA   = take((size_t)N * 128 * 2);
  size_t o_hB   = take((size_t)N * 128 * 2);
  size_t o_acc  = take((size_t)N * 128 * 4);

  float*    deg   = (float*)(ws + o_deg);
  float*    S     = (float*)(ws + o_S);
  uint32_t* cnt   = (uint32_t*)(ws + o_cnt);
  uint32_t* cur   = (uint32_t*)(ws + o_cur);
  uint32_t* ro    = (uint32_t*)(ws + o_ro);
  uint32_t* bt    = (uint32_t*)(ws + o_bt);
  uint32_t* perm  = (uint32_t*)(ws + o_perm);
  __bf16*   hA    = (__bf16*)(ws + o_hA);
  __bf16*   hB    = (__bf16*)(ws + o_hB);
  float*    acc   = (float*)(ws + o_acc);
  float*    outp  = (float*)d_out;

  long long zw = (long long)(zeroEnd / 4);
  k_zero<<<(int)((zw + 255) / 256), 256, 0, stream>>>((uint32_t*)ws, zw);
  k_count<<<(E + 255) / 256, 256, 0, stream>>>(dst, et, deg, cnt, E);
  k_prefix<<<1, 32, 0, stream>>>(cnt, ro, cur, bt);
  k_scatter<<<(E + 255) / 256, 256, 0, stream>>>(et, cur, perm, E);
  k_segS<<<(E + 255) / 256, 256, 0, stream>>>(src, dst, et, deg, S, E);
  k_layer0<<<N, 128, 0, stream>>>(S, deg, W0, Wl0, b0, hA);

  const int tiles   = N / 16;                  // 6250 node tiles (exact)
  const int slBlks  = (tiles + 3) / 4;
  const int edBlks  = E / 64 + N_RELS + 1;     // upper bound; excess blocks exit

  // Layer 1: 128 -> 128, relu
  k_selfloop<128, 128><<<slBlks, 128, 0, stream>>>(hA, Wl1, b1, acc, tiles);
  k_edge<128, 128><<<edBlks, 128, 0, stream>>>(hA, W1, src, dst, perm, ro, bt, acc);
  long long n1 = (long long)N * 128;
  k_relu_cvt<<<(int)((n1 + 255) / 256), 256, 0, stream>>>(acc, hB, n1);

  // Layer 2: 128 -> 64, no activation, straight into d_out (f32)
  k_selfloop<128, 64><<<slBlks, 128, 0, stream>>>(hB, Wl2, b2, outp, tiles);
  k_edge<128, 64><<<edBlks, 128, 0, stream>>>(hB, W2, src, dst, perm, ro, bt, outp);
}
